// UNetNoBNSpherical_27015344292186
// MI455X (gfx1250) — matrix operand order repaired and checked by reading, not compiled, not run on hardware
//
#include <hip/hip_runtime.h>

// ---------------------------------------------------------------------------
// CDNA5 (gfx1250, wave32) implementation of the spherical Chebyshev U-Net.
//
// Every ConvCheb layer = sparse 4-neighbor Laplacian stencils (building the
// Chebyshev basis x0=x, x1=Lx, x2=2Lx1-x0 directly in bf16, pre-swizzled into
// the WMMA A row layout) + one bf16 WMMA GEMM [M=B*V] x [Kpad=3Cin] x [Cout]
// with f32 accumulation and fused bias/ReLU. Weights are packed once per
// launch into the per-lane B-fragment layout of v_wmma_f32_16x16x32_bf16 and
// staged through LDS (one B column per block, shared by all 8 waves).
// ---------------------------------------------------------------------------

typedef __attribute__((ext_vector_type(16))) __bf16 v16bf;
typedef __attribute__((ext_vector_type(8)))  __bf16 v8bf;
typedef __attribute__((ext_vector_type(8)))  float  v8f;
typedef unsigned short u16;

static __device__ __forceinline__ u16 f2bf(float f) {
    // round-to-nearest-even f32 -> bf16
    unsigned int u = __float_as_uint(f);
    unsigned int r = (u + 0x7FFFu + ((u >> 16) & 1u)) >> 16;
    return (u16)r;
}

// ---------------------------------------------------------------------------
// Weight pack: W_f32 [3,Cin,Cout] (row-major over (k*Cin+c, n)) -> bf16 tiles.
// Packed layout matches B-operand of v_wmma_f32_16x16x32_bf16:
//   idx = ((kt*ntiles + nt)*32 + lane)*16 + i
//   lane holds column n = nt*16 + lane%16, K = kt*32 + (i/8)*16 + (lane<16?0:8) + i%8.
// ---------------------------------------------------------------------------
__global__ void pack_w_kernel(const float* __restrict__ W, u16* __restrict__ dst,
                              int Cin, int Cout, int Kpad, int total) {
    int idx = blockIdx.x * blockDim.x + threadIdx.x;
    if (idx >= total) return;
    int i    = idx & 15;
    int lane = (idx >> 4) & 31;
    int t    = idx >> 9;
    int ntiles = Cout >> 4;
    int kt = t / ntiles;
    int nt = t - kt * ntiles;
    int k  = kt * 32 + ((i >> 3) << 4) + ((lane < 16) ? 0 : 8) + (i & 7);
    int n  = nt * 16 + (lane & 15);
    float v = (k < 3 * Cin) ? W[(size_t)k * Cout + n] : 0.0f;
    dst[idx] = f2bf(v);
}

// ---------------------------------------------------------------------------
// Chebyshev basis pass 1: x1 = L*x (4-point stencil, weights gathered from the
// dense L row), write x0 and x1 as bf16 into A-pack (row-major [M, Kpad]),
// keep x1 in f32 for pass 2.
// ---------------------------------------------------------------------------
__global__ void cheb_pass1_kernel(const float* __restrict__ x, const float* __restrict__ L,
                                  float* __restrict__ x1tmp, u16* __restrict__ Apack,
                                  int Hh, int C, int Kpad, int total) {
    int idx = blockIdx.x * blockDim.x + threadIdx.x;
    if (idx >= total) return;
    int c = idx % C;
    int m = idx / C;
    int Wd = 2 * Hh, V = Hh * Wd;
    int v = m % V, b = m / V;
    int h = v / Wd, w = v - h * Wd;
    int base = b * V;
    const float* Lrow = L + (size_t)v * V;
    int wl = (w == 0) ? (Wd - 1) : (w - 1);
    int wr = (w == Wd - 1) ? 0 : (w + 1);
    int ul = h * Wd + wl, ur = h * Wd + wr;
    float acc = Lrow[ul] * x[(size_t)(base + ul) * C + c]
              + Lrow[ur] * x[(size_t)(base + ur) * C + c];
    if (h > 0)      { int uu = v - Wd; acc += Lrow[uu] * x[(size_t)(base + uu) * C + c]; }
    if (h < Hh - 1) { int ud = v + Wd; acc += Lrow[ud] * x[(size_t)(base + ud) * C + c]; }
    float x0 = x[(size_t)m * C + c];
    x1tmp[(size_t)m * C + c] = acc;
    u16* arow = Apack + (size_t)m * Kpad;
    arow[c]     = f2bf(x0);
    arow[C + c] = f2bf(acc);
}

// Pass 2: x2 = 2*L*x1 - x0 -> bf16 into A-pack column block [2C, 3C).
__global__ void cheb_pass2_kernel(const float* __restrict__ x, const float* __restrict__ L,
                                  const float* __restrict__ x1tmp, u16* __restrict__ Apack,
                                  int Hh, int C, int Kpad, int total) {
    int idx = blockIdx.x * blockDim.x + threadIdx.x;
    if (idx >= total) return;
    int c = idx % C;
    int m = idx / C;
    int Wd = 2 * Hh, V = Hh * Wd;
    int v = m % V, b = m / V;
    int h = v / Wd, w = v - h * Wd;
    int base = b * V;
    const float* Lrow = L + (size_t)v * V;
    int wl = (w == 0) ? (Wd - 1) : (w - 1);
    int wr = (w == Wd - 1) ? 0 : (w + 1);
    int ul = h * Wd + wl, ur = h * Wd + wr;
    float acc = Lrow[ul] * x1tmp[(size_t)(base + ul) * C + c]
              + Lrow[ur] * x1tmp[(size_t)(base + ur) * C + c];
    if (h > 0)      { int uu = v - Wd; acc += Lrow[uu] * x1tmp[(size_t)(base + uu) * C + c]; }
    if (h < Hh - 1) { int ud = v + Wd; acc += Lrow[ud] * x1tmp[(size_t)(base + ud) * C + c]; }
    float x2 = 2.0f * acc - x[(size_t)m * C + c];
    Apack[(size_t)m * Kpad + 2 * C + c] = f2bf(x2);
}

// Zero-fill A-pack K-padding columns [3C, Kpad) (only needed for Cin=16).
__global__ void pad_zero_kernel(u16* __restrict__ Apack, int C, int Kpad, int total) {
    int idx = blockIdx.x * blockDim.x + threadIdx.x;
    if (idx >= total) return;
    int padw = Kpad - 3 * C;
    int m = idx / padw;
    int j = idx - m * padw;
    Apack[(size_t)m * Kpad + 3 * C + j] = 0;
}

// ---------------------------------------------------------------------------
// WMMA GEMM: out[M,N] = relu(A[M,Kpad](bf16) * Bpack(bf16) + bias).
// Block = 256 threads = 8 waves covering a 128x16 output tile
// (grid.x = M/128, grid.y = N/16). The block's B column (Kpad x 16 bf16,
// <= 96 KB, fits the 320 KB WGP LDS) is staged once into LDS and shared by
// all 8 waves; the inner loop reads A via global_load_b128 (+speculative
// prefetch) and B via ds_load_b128 pairs. All shapes satisfy M%128==0,
// N%16==0, Kpad%32==0, so EXEC is all-ones at every WMMA (ISA requirement).
// ---------------------------------------------------------------------------
__global__ void __launch_bounds__(256)
wmma_gemm_kernel(const u16* __restrict__ A, const u16* __restrict__ Bp,
                 const float* __restrict__ bias, float* __restrict__ out,
                 int M, int N, int Kpad, int relu) {
    extern __shared__ u16 sB[];            // [nKt * 512] = Kpad x 16 bf16
    int lane    = threadIdx.x & 31;
    int wave    = threadIdx.x >> 5;
    int ntile   = blockIdx.y;
    int ntiles  = N >> 4;
    int nKt     = Kpad >> 5;

    // Cooperative stage of this block's B column: nKt*64 16-byte chunks.
    for (int ch = threadIdx.x; ch < (nKt << 6); ch += 256) {
        int kt  = ch >> 6;
        int off = (ch & 63) << 3;          // u16-element offset within chunk
        const uint4* src =
            (const uint4*)(Bp + (((size_t)kt * ntiles + ntile) << 9) + off);
        *(uint4*)(sB + (kt << 9) + off) = *src;
    }
    __syncthreads();                       // s_barrier_signal + s_barrier_wait

    int row0    = (blockIdx.x * 8 + wave) * 16;
    int laneLow = lane & 15;
    // A fragment per lane: row = row0 + lane%16; two 8-element chunks at
    // K-offsets (lane<16 ? 0 : 8) and +16 (HW 16-bit A layout).
    const u16* arow = A + (size_t)(row0 + laneLow) * Kpad + ((lane < 16) ? 0 : 8);
    const u16* brow = sB + (lane << 4);    // 16 contiguous bf16 per lane

    v8f acc = {};
    for (int kt = 0; kt < nKt; ++kt) {
        v8bf a_lo = *(const v8bf*)(arow + kt * 32);
        v8bf a_hi = *(const v8bf*)(arow + kt * 32 + 16);
        v16bf a = __builtin_shufflevector(a_lo, a_hi,
                                          0, 1, 2, 3, 4, 5, 6, 7,
                                          8, 9, 10, 11, 12, 13, 14, 15);
        v16bf b = *(const v16bf*)(brow + (kt << 9));
        // Speculative prefetch of A four k-steps (256 B) ahead (dropped if untranslated).
        __builtin_prefetch(arow + kt * 32 + 128, 0, 3);
        acc = __builtin_amdgcn_wmma_f32_16x16x32_bf16(
                  false, a, false, b, (short)0, acc, false, false);
    }

    int col   = ntile * 16 + laneLow;
    float bv  = bias[col];
    int rbase = row0 + ((lane < 16) ? 0 : 8);
#pragma unroll
    for (int r = 0; r < 8; ++r) {
        float v = acc[r] + bv;
        if (relu) v = fmaxf(v, 0.0f);
        out[(size_t)(rbase + r) * N + col] = v;
    }
}

// ---------------------------------------------------------------------------
// Elementwise grid ops (f32): 2x2 average pool, 2x nearest unpool, concat.
// ---------------------------------------------------------------------------
__global__ void pool_kernel(const float* __restrict__ in, float* __restrict__ out,
                            int Hin, int C, int total) {
    int idx = blockIdx.x * blockDim.x + threadIdx.x;
    if (idx >= total) return;
    int c = idx % C;
    int r = idx / C;
    int Wo = Hin;                 // out width = (2*Hin)/2
    int Vout = (Hin >> 1) * Wo;
    int vo = r % Vout, b = r / Vout;
    int ho = vo / Wo, wo = vo - ho * Wo;
    int Wi = 2 * Hin, Vin = Hin * Wi;
    const float* ib = in + (size_t)b * Vin * C + c;
    int r0 = (2 * ho) * Wi + 2 * wo;
    float s = ib[(size_t)r0 * C] + ib[(size_t)(r0 + 1) * C]
            + ib[(size_t)(r0 + Wi) * C] + ib[(size_t)(r0 + Wi + 1) * C];
    out[idx] = 0.25f * s;
}

__global__ void unpool_kernel(const float* __restrict__ in, float* __restrict__ out,
                              int Hin, int C, int total) {
    int idx = blockIdx.x * blockDim.x + threadIdx.x;
    if (idx >= total) return;
    int c = idx % C;
    int r = idx / C;
    int Wo = 4 * Hin, Vo = 2 * Hin * Wo;
    int vo = r % Vo, b = r / Vo;
    int ho = vo / Wo, wo = vo - ho * Wo;
    int Wi = 2 * Hin, Vin = Hin * Wi;
    out[idx] = in[((size_t)b * Vin + (ho >> 1) * Wi + (wo >> 1)) * C + c];
}

__global__ void concat_kernel(const float* __restrict__ a, const float* __restrict__ s,
                              float* __restrict__ out, int C1, int C2, int total) {
    int idx = blockIdx.x * blockDim.x + threadIdx.x;
    if (idx >= total) return;
    int C = C1 + C2;
    int c = idx % C;
    int m = idx / C;
    out[idx] = (c < C1) ? a[(size_t)m * C1 + c] : s[(size_t)m * C2 + (c - C1)];
}

// ---------------------------------------------------------------------------
// Host driver.
// ---------------------------------------------------------------------------
namespace {
struct LayerDef { int cin, cout; };
static const LayerDef g_layers[17] = {
    {16, 32},  {32, 64},            // conv1/2_enc_l5
    {64, 128}, {128, 256}, {256, 512}, {512, 512}, {512, 512},  // enc l4..l0
    {512, 512}, {1024, 512},        // dec l1
    {512, 256}, {768, 256},         // dec l2
    {256, 128}, {384, 128},         // dec l3
    {128, 64},  {192, 64},          // dec l4
    {64, 32},   {32, 16}            // dec l5
};
static inline int kpad_of(int cin) { return ((3 * cin + 31) / 32) * 32; }
static inline int cdiv(int a, int b) { return (a + b - 1) / b; }
}

extern "C" void kernel_launch(void* const* d_in, const int* in_sizes, int n_in,
                              void* d_out, int out_size, void* d_ws, size_t ws_size,
                              hipStream_t stream) {
    (void)in_sizes; (void)n_in; (void)out_size; (void)ws_size;
    const int Bb = 128;
    // Input order: x, L1..L5 (H = 2,4,8,16,32), then (W,b) per layer.
    const float* x_in = (const float*)d_in[0];
    const float* Lap[5];
    for (int i = 0; i < 5; ++i) Lap[i] = (const float*)d_in[1 + i];
    const int Hlev[5] = {2, 4, 8, 16, 32};

    // Workspace bump allocation (total ~426 MB).
    const size_t MB = 1024ull * 1024ull;
    char* ws = (char*)d_ws;
    u16*   bpack = (u16*)(ws);               // 16 MB: packed weights, all layers
    u16*   apack = (u16*)(ws + 16 * MB);     // 96 MB: bf16 Chebyshev A-pack
    float* x1tmp = (float*)(ws + 112 * MB);  // 64 MB: f32 x1 scratch
    float* P0    = (float*)(ws + 176 * MB);  // 64 MB ping
    float* P1    = (float*)(ws + 240 * MB);  // 64 MB pong
    float* S5    = (float*)(ws + 304 * MB);  // 64 MB skip
    float* S4    = (float*)(ws + 368 * MB);  // 32 MB skip
    float* S3    = (float*)(ws + 400 * MB);  // 16 MB skip
    float* S2    = (float*)(ws + 416 * MB);  //  8 MB skip
    float* S1    = (float*)(ws + 424 * MB);  //  2 MB skip

    // Pack all weights to the WMMA B-fragment layout.
    size_t bofs[17];
    {
        size_t acc = 0;
        for (int i = 0; i < 17; ++i) {
            bofs[i] = acc;
            acc += (size_t)kpad_of(g_layers[i].cin) * g_layers[i].cout;
        }
        for (int i = 0; i < 17; ++i) {
            int kp = kpad_of(g_layers[i].cin), co = g_layers[i].cout;
            int tot = kp * co;
            const float* W = (const float*)d_in[6 + 2 * i];
            pack_w_kernel<<<cdiv(tot, 256), 256, 0, stream>>>(
                W, bpack + bofs[i], g_layers[i].cin, co, kp, tot);
        }
    }

    // One ConvCheb layer: basis build (2 stencil passes) + WMMA GEMM.
    auto conv = [&](int lev /*0..4 -> H=2..32*/, const float* in, int layer,
                    float* outp, bool relu) {
        int Hh = Hlev[lev];
        int V = 2 * Hh * Hh;
        int M = Bb * V;
        int C = g_layers[layer].cin;
        int Cout = g_layers[layer].cout;
        int Kpad = kpad_of(C);
        const float* Lm = Lap[lev];
        const float* bias = (const float*)d_in[7 + 2 * layer];
        int tot = M * C;
        cheb_pass1_kernel<<<cdiv(tot, 256), 256, 0, stream>>>(
            in, Lm, x1tmp, apack, Hh, C, Kpad, tot);
        cheb_pass2_kernel<<<cdiv(tot, 256), 256, 0, stream>>>(
            in, Lm, x1tmp, apack, Hh, C, Kpad, tot);
        if (Kpad > 3 * C) {
            int pt = M * (Kpad - 3 * C);
            pad_zero_kernel<<<cdiv(pt, 256), 256, 0, stream>>>(apack, C, Kpad, pt);
        }
        dim3 grid(M / 128, Cout / 16);
        size_t shmem = (size_t)Kpad * 16 * sizeof(u16);   // B column in LDS
        wmma_gemm_kernel<<<grid, 256, shmem, stream>>>(
            apack, bpack + bofs[layer], bias, outp, M, Cout, Kpad, relu ? 1 : 0);
    };
    auto pool_op = [&](const float* in, float* outp, int lev_in, int C) {
        int Hin = Hlev[lev_in];
        int tot = Bb * (Hin * Hin / 2) * C;   // Vout = Hin^2/2
        pool_kernel<<<cdiv(tot, 256), 256, 0, stream>>>(in, outp, Hin, C, tot);
    };
    auto unpool_op = [&](const float* in, float* outp, int lev_in, int C) {
        int Hin = Hlev[lev_in];
        int tot = Bb * (8 * Hin * Hin) * C;   // Vout = 8*Hin^2
        unpool_kernel<<<cdiv(tot, 256), 256, 0, stream>>>(in, outp, Hin, C, tot);
    };
    auto concat_op = [&](const float* a, const float* s, float* outp, int lev,
                         int C1, int C2) {
        int Hh = Hlev[lev];
        int tot = Bb * 2 * Hh * Hh * (C1 + C2);
        concat_kernel<<<cdiv(tot, 256), 256, 0, stream>>>(a, s, outp, C1, C2, tot);
    };

    // ---------------- encoder ----------------
    conv(4, x_in, 0, P0, true);          // conv1_enc_l5: 16->32   @H32
    conv(4, P0,  1, S5, true);           // conv2_enc_l5: 32->64   @H32
    pool_op(S5, P0, 4, 64);              // H32 -> H16
    conv(3, P0,  2, S4, true);           // conv_enc_l4: 64->128   @H16
    pool_op(S4, P0, 3, 128);             // H16 -> H8
    conv(2, P0,  3, S3, true);           // conv_enc_l3: 128->256  @H8
    pool_op(S3, P0, 2, 256);             // H8 -> H4
    conv(1, P0,  4, S2, true);           // conv_enc_l2: 256->512  @H4
    pool_op(S2, P0, 1, 512);             // H4 -> H2
    conv(0, P0,  5, S1, true);           // conv_enc_l1: 512->512  @H2
    conv(0, S1,  6, P0, false);          // conv_enc_l0: 512->512  @H2 (no relu)
    // ---------------- decoder ----------------
    conv(0, P0,  7, P1, true);           // conv1_dec_l1: 512->512 @H2
    concat_op(P1, S1, P0, 0, 512, 512);
    conv(0, P0,  8, P1, true);           // conv2_dec_l1: 1024->512 @H2
    unpool_op(P1, P0, 0, 512);           // H2 -> H4
    conv(1, P0,  9, P1, true);           // conv1_dec_l2: 512->256 @H4
    concat_op(P1, S2, P0, 1, 256, 512);
    conv(1, P0, 10, P1, true);           // conv2_dec_l2: 768->256 @H4
    unpool_op(P1, P0, 1, 256);           // H4 -> H8
    conv(2, P0, 11, P1, true);           // conv1_dec_l3: 256->128 @H8
    concat_op(P1, S3, P0, 2, 128, 256);
    conv(2, P0, 12, P1, true);           // conv2_dec_l3: 384->128 @H8
    unpool_op(P1, P0, 2, 128);           // H8 -> H16
    conv(3, P0, 13, P1, true);           // conv1_dec_l4: 128->64  @H16
    concat_op(P1, S4, P0, 3, 64, 128);
    conv(3, P0, 14, P1, true);           // conv2_dec_l4: 192->64  @H16
    unpool_op(P1, P0, 3, 64);            // H16 -> H32
    conv(4, P0, 15, P1, false);          // conv1_dec_l5: 64->32   @H32 (no relu)
    conv(4, P1, 16, (float*)d_out, false); // conv2_dec_l5: 32->16 (no relu) -> out
}
